// ConditionalVQVAE_embeddingSpaceNet_13477607374899
// MI455X (gfx1250) — matrix operand order repaired and checked by reading, not compiled, not run on hardware
//
#include <hip/hip_runtime.h>

// ---------------------------------------------------------------------------
// VQ-VAE nearest-code lookup for MI455X (gfx1250, wave32, WMMA).
//   argmin_n ||ze - d_n||^2  ==  argmin_n ( 0.5*||d_n||^2 - ze . d_n )
// Cross term computed as bf16 split GEMM (hi/lo, 3 WMMAs per K=32) with f32
// accumulation on v_wmma_f32_16x16x32_bf16 — ~f32-accurate, 2.7x fewer matrix
// instructions than the 16x16x4 f32 WMMA path.
// ---------------------------------------------------------------------------

typedef __bf16 bf16_t;
typedef __attribute__((ext_vector_type(16))) __bf16 v16bf;
typedef __attribute__((ext_vector_type(8)))  __bf16 v8bf;
typedef __attribute__((ext_vector_type(8)))  float  v8f;
typedef __attribute__((ext_vector_type(4)))  float  fvec4;

constexpr int D_DIM    = 256;            // feature dim (K of the GEMM)
constexpr int N_CODES  = 1024;           // codebook size
constexpr int M_TOTAL  = 32 * 2048;      // B*T query rows
constexpr int ROWS_PER_BLOCK = 128;      // 8 waves * 16 rows
constexpr int LDS_ROW_STRIDE = 528;      // 512B of bf16 K-data + 16B pad (bank skew)

// --------------------------- pre-pass -------------------------------------
// Split dictionary rows into bf16 hi/lo planes and compute 0.5*||d_n||^2.
__global__ __launch_bounds__(256)
void vq_prep_kernel(const float* __restrict__ dict,
                    bf16_t* __restrict__ dhi,
                    bf16_t* __restrict__ dlo,
                    float*  __restrict__ halfsq) {
  const int n = blockIdx.x;      // code index
  const int d = threadIdx.x;     // feature index
  const float x = dict[n * D_DIM + d];
  const bf16_t h = (bf16_t)x;
  const float  hf = (float)h;
  const bf16_t l = (bf16_t)(x - hf);
  dhi[n * D_DIM + d] = h;
  dlo[n * D_DIM + d] = l;

  __shared__ float red[256];
  red[d] = x * x;
  __syncthreads();
  #pragma unroll
  for (int s = 128; s > 0; s >>= 1) {
    if (d < s) red[d] += red[d + s];
    __syncthreads();
  }
  if (d == 0) halfsq[n] = 0.5f * red[0];
}

// --------------------------- main kernel ----------------------------------
__global__ __launch_bounds__(256)
void vq_main_kernel(const float* __restrict__ ze,
                    const float* __restrict__ dict,
                    const bf16_t* __restrict__ dhi,
                    const bf16_t* __restrict__ dlo,
                    const float* __restrict__ halfsq,
                    float* __restrict__ out) {
  // B tile staging: [term(hi/lo)][16 rows][512B K-data + 16B pad]
  __shared__ __align__(16) unsigned char ldsB[2 * 16 * LDS_ROW_STRIDE];

  const int tid  = threadIdx.x;
  const int wave = tid >> 5;
  const int lane = tid & 31;
  const int l15  = lane & 15;        // M row within wave tile / N within chunk
  const int lhi  = lane >> 4;        // lane half (selects K sub-range)

  const int rowBase = blockIdx.x * ROWS_PER_BLOCK + wave * 16;
  const int rowM    = rowBase + l15;

  // ---- Build A operand (16 rows x K=256, bf16 hi/lo split) in registers ----
  // ISA 16-bit A 16x32 layout: lanes 0-15 hold K=[k0..k0+7]+[k0+16..k0+23],
  // lanes 16-31 hold K=[k0+8..k0+15]+[k0+24..k0+31].
  v16bf a_hi[8], a_lo[8];
  {
    const float* zrow = ze + (size_t)rowM * D_DIM;
    #pragma unroll
    for (int kt = 0; kt < 8; ++kt) {
      const int k0 = kt * 32;
      const int s0 = k0 + 8 * lhi;        // first 8-element K segment
      const int s1 = k0 + 16 + 8 * lhi;   // second 8-element K segment
      fvec4 p0 = *(const fvec4*)(zrow + s0);
      fvec4 p1 = *(const fvec4*)(zrow + s0 + 4);
      fvec4 p2 = *(const fvec4*)(zrow + s1);
      fvec4 p3 = *(const fvec4*)(zrow + s1 + 4);
      v16bf h, l;
      #pragma unroll
      for (int i = 0; i < 4; ++i) {
        float x;
        bf16_t hh;
        x = p0[i]; hh = (bf16_t)x; h[i]      = hh; l[i]      = (bf16_t)(x - (float)hh);
        x = p1[i]; hh = (bf16_t)x; h[4 + i]  = hh; l[4 + i]  = (bf16_t)(x - (float)hh);
        x = p2[i]; hh = (bf16_t)x; h[8 + i]  = hh; l[8 + i]  = (bf16_t)(x - (float)hh);
        x = p3[i]; hh = (bf16_t)x; h[12 + i] = hh; l[12 + i] = (bf16_t)(x - (float)hh);
      }
      a_hi[kt] = h;
      a_lo[kt] = l;
    }
  }

  // Running argmin state: VGPR r of the accumulator maps to row (rowBase+r)
  // for lanes 0-15 and row (rowBase+8+r) for lanes 16-31; N = n0 + l15.
  float mins[8];
  int   midx[8];
  #pragma unroll
  for (int r = 0; r < 8; ++r) { mins[r] = 3.0e38f; midx[r] = 0; }

  for (int nc = 0; nc < N_CODES / 16; ++nc) {
    const int n0 = nc * 16;

    // ---- Stage 16-code dict tile (hi+lo) into LDS, padded rows ----
    {
      const int r = tid >> 4;   // 0..15 row
      const int j = tid & 15;   // 0..15 : 32B slot
      const size_t gsrc = (size_t)(n0 + r) * (D_DIM * 2) + (size_t)j * 32;
      const fvec4* srcHi = (const fvec4*)((const unsigned char*)dhi + gsrc);
      const fvec4* srcLo = (const fvec4*)((const unsigned char*)dlo + gsrc);
      fvec4* dstHi = (fvec4*)(ldsB + r * LDS_ROW_STRIDE + j * 32);
      fvec4* dstLo = (fvec4*)(ldsB + 16 * LDS_ROW_STRIDE + r * LDS_ROW_STRIDE + j * 32);
      dstHi[0] = srcHi[0]; dstHi[1] = srcHi[1];
      dstLo[0] = srcLo[0]; dstLo[1] = srcLo[1];
    }
    __syncthreads();

    // ---- 16x16 cross tile via WMMA, K=256 in 8 steps of 32 ----
    // ISA 16-bit B 32x16 layout: lanes 0-15 hold K=[k0..k0+7]+[k0+16..k0+23],
    // lanes 16-31 hold K=[k0+8..k0+15]+[k0+24..k0+31] (matches A).
    v8f c = {};
    const unsigned char* bHiRow = ldsB + l15 * LDS_ROW_STRIDE;
    const unsigned char* bLoRow = bHiRow + 16 * LDS_ROW_STRIDE;
    #pragma unroll
    for (int kt = 0; kt < 8; ++kt) {
      const int kb = (kt * 32 + 8 * lhi) * 2;   // byte offset of K segment
      v8bf bh0 = *(const v8bf*)(bHiRow + kb);
      v8bf bh1 = *(const v8bf*)(bHiRow + kb + 32);
      v8bf bl0 = *(const v8bf*)(bLoRow + kb);
      v8bf bl1 = *(const v8bf*)(bLoRow + kb + 32);
      v16bf bh = __builtin_shufflevector(bh0, bh1, 0,1,2,3,4,5,6,7,8,9,10,11,12,13,14,15);
      v16bf bl = __builtin_shufflevector(bl0, bl1, 0,1,2,3,4,5,6,7,8,9,10,11,12,13,14,15);
      // cross ≈ Ahi*Bhi + Ahi*Blo + Alo*Bhi  (lo*lo term ~2^-18, negligible)
      c = __builtin_amdgcn_wmma_f32_16x16x32_bf16(false, a_hi[kt], false, bh,
                                                  (short)0, c, false, false);
      c = __builtin_amdgcn_wmma_f32_16x16x32_bf16(false, a_hi[kt], false, bl,
                                                  (short)0, c, false, false);
      c = __builtin_amdgcn_wmma_f32_16x16x32_bf16(false, a_lo[kt], false, bh,
                                                  (short)0, c, false, false);
    }
    __syncthreads();

    // ---- score = 0.5*||d_n||^2 - cross ; running argmin ----
    const int   n   = n0 + l15;
    const float hsq = halfsq[n];
    #pragma unroll
    for (int r = 0; r < 8; ++r) {
      const float s = hsq - c[r];
      if (s < mins[r]) { mins[r] = s; midx[r] = n; }   // strict < keeps lowest n
    }
  }

  // ---- Reduce across the 16 lanes of each half (N%16 classes) ----
  #pragma unroll
  for (int r = 0; r < 8; ++r) {
    float s  = mins[r];
    int   id = midx[r];
    #pragma unroll
    for (int m = 1; m <= 8; m <<= 1) {     // xor<=8 stays within the 16-lane half
      const float so = __shfl_xor(s, m, 32);
      const int   io = __shfl_xor(id, m, 32);
      if (so < s || (so == s && io < id)) { s = so; id = io; }
    }
    mins[r] = s;
    midx[r] = id;
  }

  // ---- Gather winning f32 dictionary rows: 16 rows per wave, 32-lane copy ----
  #pragma unroll
  for (int r = 0; r < 8; ++r) {
    const int z0 = __shfl(midx[r], 0, 32);   // row rowBase + r
    const int z1 = __shfl(midx[r], 16, 32);  // row rowBase + 8 + r
    const fvec4* s0 = (const fvec4*)(dict + (size_t)z0 * D_DIM) + lane * 2;
    const fvec4* s1 = (const fvec4*)(dict + (size_t)z1 * D_DIM) + lane * 2;
    fvec4* d0 = (fvec4*)(out + (size_t)(rowBase + r) * D_DIM) + lane * 2;
    fvec4* d1 = (fvec4*)(out + (size_t)(rowBase + 8 + r) * D_DIM) + lane * 2;
    d0[0] = s0[0]; d0[1] = s0[1];
    d1[0] = s1[0]; d1[1] = s1[1];
  }
}

// --------------------------- host launch ----------------------------------
extern "C" void kernel_launch(void* const* d_in, const int* in_sizes, int n_in,
                              void* d_out, int out_size, void* d_ws, size_t ws_size,
                              hipStream_t stream) {
  (void)in_sizes; (void)n_in; (void)out_size; (void)ws_size;
  const float* ze   = (const float*)d_in[0];   // [32,2048,256] f32
  const float* dict = (const float*)d_in[1];   // [1024,256]   f32
  float* out = (float*)d_out;                  // [32,2048,256] f32

  // workspace: dict_hi bf16 (512KB) | dict_lo bf16 (512KB) | halfsq f32 (4KB)
  unsigned char* ws = (unsigned char*)d_ws;
  bf16_t* dhi    = (bf16_t*)(ws);
  bf16_t* dlo    = (bf16_t*)(ws + 512 * 1024);
  float*  halfsq = (float*)(ws + 1024 * 1024);

  vq_prep_kernel<<<N_CODES, 256, 0, stream>>>(dict, dhi, dlo, halfsq);
  vq_main_kernel<<<M_TOTAL / ROWS_PER_BLOCK, 256, 0, stream>>>(
      ze, dict, dhi, dlo, halfsq, out);
}
